// JointMaximumMeanDiscrepancyLoss_37615323578679
// MI455X (gfx1250) — matrix operand order repaired and checked by reading, not compiled, not run on hardware
//
#include <hip/hip_runtime.h>
#include <math.h>

#define NROW   8192
#define HALF_B 4096
#define DIM0   256
#define DIM1   64
#define KTOT   320
#define NSTEP  10           // KTOT / 32
#define NT     64           // 8192/128 tiles per side

typedef __attribute__((ext_vector_type(16))) __bf16 v16bf;
typedef __attribute__((ext_vector_type(8)))  float  v8f;

struct WS {
  float  sqA[NROW];
  float  sqB[NROW];
  float  a[NROW];
  float  colp0[32 * DIM0];
  float  colp1[32 * DIM1];
  float  gamma0, gamma1, sg0, sg1;
  double partial[NT * NT];
};

union Frag   { uint4 q[2]; v16bf v; };
union Pack16 { unsigned short h[16]; uint4 q[2]; };

__device__ __forceinline__ unsigned bf16_rne(float x) {
  unsigned u = __float_as_uint(x);
  return (u + 0x7FFFu + ((u >> 16) & 1u)) >> 16;
}

// ---------------- Kernel 1: per-row squared norms (1 wave / row) ----------------
__global__ __launch_bounds__(256) void k_rownorm(const float* __restrict__ s0,
                                                 const float* __restrict__ s1,
                                                 const float* __restrict__ t0,
                                                 const float* __restrict__ t1,
                                                 WS* __restrict__ ws) {
  int wave = threadIdx.x >> 5, lane = threadIdx.x & 31;
  int row  = blockIdx.x * 8 + wave;
  const float* f0 = (row < HALF_B) ? (s0 + (size_t)row * DIM0)
                                   : (t0 + (size_t)(row - HALF_B) * DIM0);
  const float* f1 = (row < HALF_B) ? (s1 + (size_t)row * DIM1)
                                   : (t1 + (size_t)(row - HALF_B) * DIM1);
  float q0 = 0.f, q1 = 0.f;
#pragma unroll
  for (int i = 0; i < DIM0; i += 32) { float x = f0[i + lane]; q0 += x * x; }
#pragma unroll
  for (int i = 0; i < DIM1; i += 32) { float x = f1[i + lane]; q1 += x * x; }
#pragma unroll
  for (int m = 16; m >= 1; m >>= 1) {
    q0 += __shfl_xor(q0, m, 32);
    q1 += __shfl_xor(q1, m, 32);
  }
  if (lane == 0) { ws->sqA[row] = q0; ws->sqB[row] = q1; }
}

// ---------------- Kernel 2: per-strip column sums (deterministic, no atomics) ---
__global__ __launch_bounds__(256) void k_colsum(const float* __restrict__ s0,
                                                const float* __restrict__ s1,
                                                const float* __restrict__ t0,
                                                const float* __restrict__ t1,
                                                WS* __restrict__ ws) {
  int c  = threadIdx.x;
  int b  = blockIdx.x;                 // 0..31, strip of 256 rows
  int r0 = b * 256;
  const float* f0 = (r0 < HALF_B) ? (s0 + (size_t)r0 * DIM0)
                                  : (t0 + (size_t)(r0 - HALF_B) * DIM0);
  const float* f1 = (r0 < HALF_B) ? (s1 + (size_t)r0 * DIM1)
                                  : (t1 + (size_t)(r0 - HALF_B) * DIM1);
  float p0 = 0.f;
  for (int r = 0; r < 256; ++r) p0 += f0[(size_t)r * DIM0 + c];
  ws->colp0[b * DIM0 + c] = p0;
  if (c < DIM1) {
    float p1 = 0.f;
    for (int r = 0; r < 256; ++r) p1 += f1[(size_t)r * DIM1 + c];
    ws->colp1[b * DIM1 + c] = p1;
  }
}

// ---------------- Kernel 3: gammas (f64 scalar math) + a[] --------------------
__global__ __launch_bounds__(256) void k_gamma(WS* __restrict__ ws) {
  __shared__ double red[256];
  __shared__ double res[4];
  __shared__ float  gsh[2];
  int t = threadIdx.x;
  double ssq0 = 0.0, ssq1 = 0.0;
  for (int i = t; i < NROW; i += 256) {
    ssq0 += (double)ws->sqA[i];
    ssq1 += (double)ws->sqB[i];
  }
  double cs0 = 0.0, cs1 = 0.0;
  {
    float c = 0.f;
    for (int b = 0; b < 32; ++b) c += ws->colp0[b * DIM0 + t];
    cs0 = (double)c * (double)c;
  }
  if (t < DIM1) {
    float c = 0.f;
    for (int b = 0; b < 32; ++b) c += ws->colp1[b * DIM1 + t];
    cs1 = (double)c * (double)c;
  }
  double vals[4] = {ssq0, ssq1, cs0, cs1};
  for (int q = 0; q < 4; ++q) {
    red[t] = vals[q];
    __syncthreads();
    for (int s = 128; s >= 1; s >>= 1) {
      if (t < s) red[t] += red[t + s];
      __syncthreads();
    }
    if (t == 0) res[q] = red[0];
    __syncthreads();
  }
  if (t == 0) {
    double n   = (double)NROW;
    double num = n * n - n;
    double g0  = num / (2.0 * n * res[0] - 2.0 * res[2]);
    double g1  = num / (2.0 * n * res[1] - 2.0 * res[3]);
    ws->gamma0 = (float)g0; ws->gamma1 = (float)g1;
    ws->sg0 = sqrtf((float)g0); ws->sg1 = sqrtf((float)g1);
    gsh[0] = (float)g0; gsh[1] = (float)g1;
  }
  __syncthreads();
  float g0 = gsh[0], g1 = gsh[1];
  for (int i = t; i < NROW; i += 256)
    ws->a[i] = g0 * ws->sqA[i] + g1 * ws->sqB[i];
}

// ---------------- Kernel 4: fused WMMA Gram + exp + weighted reduce -----------
// 128x128 block tile, upper-triangle only; split-bf16 (hi+lo) for ~f32 accuracy.
// Double-buffered LDS + register prefetch: 1 barrier / K-step, global latency
// hidden behind the WMMA phase.
#define LROW  40                 // bf16 per LDS row: 32 + 8 pad (bank-conflict free)
#define TSZ   (128 * LROW)       // one buffer, in ushorts

__device__ __forceinline__ void cvt_store16(const float4* x4, float scale,
                                            unsigned short* dhi,
                                            unsigned short* dlo) {
  float x[16] = {x4[0].x, x4[0].y, x4[0].z, x4[0].w,
                 x4[1].x, x4[1].y, x4[1].z, x4[1].w,
                 x4[2].x, x4[2].y, x4[2].z, x4[2].w,
                 x4[3].x, x4[3].y, x4[3].z, x4[3].w};
  Pack16 ph, pl;
#pragma unroll
  for (int e = 0; e < 16; ++e) {
    float xs = x[e] * scale;
    unsigned hb = bf16_rne(xs);
    float hf = __uint_as_float(hb << 16);
    pl.h[e] = (unsigned short)bf16_rne(xs - hf);
    ph.h[e] = (unsigned short)hb;
  }
  ((uint4*)dhi)[0] = ph.q[0]; ((uint4*)dhi)[1] = ph.q[1];
  ((uint4*)dlo)[0] = pl.q[0]; ((uint4*)dlo)[1] = pl.q[1];
}

__global__ __launch_bounds__(256) void k_mmd(const float* __restrict__ s0,
                                             const float* __restrict__ s1,
                                             const float* __restrict__ t0,
                                             const float* __restrict__ t1,
                                             WS* __restrict__ ws) {
  __shared__ unsigned short sAh[2 * TSZ];
  __shared__ unsigned short sAl[2 * TSZ];
  __shared__ unsigned short sBh[2 * TSZ];
  __shared__ unsigned short sBl[2 * TSZ];

  int tj = blockIdx.x, ti = blockIdx.y;
  int pidx = ti * NT + tj;
  if (tj < ti) {                       // uniform block-level cull: EXEC stays full
    if (threadIdx.x == 0) ws->partial[pidx] = 0.0;
    return;
  }
  int rowBase = ti * 128, colBase = tj * 128;
  int tid  = threadIdx.x;
  int wave = tid >> 5, lane = tid & 31;
  int wm = wave >> 2, wn = wave & 3;   // 2x4 wave grid -> 64x32 per wave
  int lg = lane >> 4, ln = lane & 15;

  float sg0 = ws->sg0, sg1 = ws->sg1;

  v8f acc[4][2];
#pragma unroll
  for (int mi = 0; mi < 4; ++mi)
#pragma unroll
    for (int ni = 0; ni < 2; ++ni)
#pragma unroll
      for (int e = 0; e < 8; ++e) acc[mi][ni][e] = 0.f;

  int lr = tid >> 1;                   // 0..127: tile row this thread stages
  int lh = (tid & 1) << 4;             // 0 or 16: which K half

  // src pointer for K-offset k (uniform control: K block never straddles 256)
  auto srcFor = [&](int k, int base, float& scl) -> const float* {
    bool m0 = (k < DIM0);
    const float* sb = m0 ? s0 : s1;
    const float* tb = m0 ? t0 : t1;
    int Dd = m0 ? DIM0 : DIM1;
    int kk = m0 ? k : (k - DIM0);
    scl = m0 ? sg0 : sg1;
    return ((base < HALF_B) ? sb : tb) +
           (size_t)((base & (HALF_B - 1)) + lr) * Dd + kk + lh;
  };

  // prologue: prefetch K-step 0 into registers
  float4 pa[4], pb[4];
  float curScale;
  {
    float scl;
    const float4* a4 = (const float4*)srcFor(0, rowBase, scl);
    const float4* b4 = (const float4*)srcFor(0, colBase, scl);
    pa[0] = a4[0]; pa[1] = a4[1]; pa[2] = a4[2]; pa[3] = a4[3];
    pb[0] = b4[0]; pb[1] = b4[1]; pb[2] = b4[2]; pb[3] = b4[3];
    curScale = scl;
  }

  for (int ks = 0; ks < NSTEP; ++ks) {
    int boff = (ks & 1) * TSZ;
    unsigned short* Ah = sAh + boff;
    unsigned short* Al = sAl + boff;
    unsigned short* Bh = sBh + boff;
    unsigned short* Bl = sBl + boff;

    // stage current slice (regs -> LDS), split f32 into bf16 hi/lo
    cvt_store16(pa, curScale, &Ah[lr * LROW + lh], &Al[lr * LROW + lh]);
    cvt_store16(pb, curScale, &Bh[lr * LROW + lh], &Bl[lr * LROW + lh]);

    // prefetch next slice into registers (in flight across barrier + WMMAs)
    if (ks + 1 < NSTEP) {
      float scl;
      const float4* a4 = (const float4*)srcFor((ks + 1) * 32, rowBase, scl);
      const float4* b4 = (const float4*)srcFor((ks + 1) * 32, colBase, scl);
      pa[0] = a4[0]; pa[1] = a4[1]; pa[2] = a4[2]; pa[3] = a4[3];
      pb[0] = b4[0]; pb[1] = b4[1]; pb[2] = b4[2]; pb[3] = b4[3];
      curScale = scl;
    }

    __syncthreads();   // single barrier per step (double buffer fences re-store)

    // A frag (ISA 7.12.2): lane ln = row M; lg=0 -> K{0..7,16..23}, lg=1 -> K{8..15,24..31}
    Frag fAh[4], fAl[4], fBh[2], fBl[2];
#pragma unroll
    for (int mi = 0; mi < 4; ++mi) {
      int rr = wm * 64 + mi * 16 + ln;
      const unsigned short* p  = &Ah[rr * LROW + lg * 8];
      fAh[mi].q[0] = *(const uint4*)p;
      fAh[mi].q[1] = *(const uint4*)(p + 16);
      const unsigned short* p2 = &Al[rr * LROW + lg * 8];
      fAl[mi].q[0] = *(const uint4*)p2;
      fAl[mi].q[1] = *(const uint4*)(p2 + 16);
    }
    // B frag: lane ln = col N; lg=0 -> K{0..7,16..23}, lg=1 -> K{8..15,24..31}
#pragma unroll
    for (int ni = 0; ni < 2; ++ni) {
      int rr = wn * 32 + ni * 16 + ln;
      const unsigned short* p  = &Bh[rr * LROW + lg * 8];
      fBh[ni].q[0] = *(const uint4*)p;
      fBh[ni].q[1] = *(const uint4*)(p + 16);
      const unsigned short* p2 = &Bl[rr * LROW + lg * 8];
      fBl[ni].q[0] = *(const uint4*)p2;
      fBl[ni].q[1] = *(const uint4*)(p2 + 16);
    }
    // split-bf16 Gram: hi*hi + hi*lo + lo*hi (lo*lo ~2^-32, dropped)
#pragma unroll
    for (int mi = 0; mi < 4; ++mi)
#pragma unroll
      for (int ni = 0; ni < 2; ++ni) {
        acc[mi][ni] = __builtin_amdgcn_wmma_f32_16x16x32_bf16(
            false, fAh[mi].v, false, fBh[ni].v, (short)0, acc[mi][ni], false, false);
        acc[mi][ni] = __builtin_amdgcn_wmma_f32_16x16x32_bf16(
            false, fAh[mi].v, false, fBl[ni].v, (short)0, acc[mi][ni], false, false);
        acc[mi][ni] = __builtin_amdgcn_wmma_f32_16x16x32_bf16(
            false, fAl[mi].v, false, fBh[ni].v, (short)0, acc[mi][ni], false, false);
      }
  }

  // Epilogue: K(i,j) = exp(2*Gs - a_i - a_j); C/D layout: M = v + 8*lg, N = ln
  const float* aArr = ws->a;
  float local = 0.f;
#pragma unroll
  for (int ni = 0; ni < 2; ++ni) {
    int j = colBase + wn * 32 + ni * 16 + ln;
    float aj = aArr[j];
#pragma unroll
    for (int mi = 0; mi < 4; ++mi) {
      int ibase = rowBase + wm * 64 + mi * 16 + lg * 8;
#pragma unroll
      for (int v = 0; v < 8; ++v) {
        float ai = aArr[ibase + v];
        local += __expf(2.f * acc[mi][ni][v] - ai - aj);
      }
    }
  }
  __syncthreads();                      // all frag reads done before LDS reuse
  float* red = (float*)sAh;
  red[tid] = local;
  __syncthreads();
  for (int s = 128; s >= 1; s >>= 1) {
    if (tid < s) red[tid] += red[tid + s];
    __syncthreads();
  }
  if (tid == 0) {
    float wsym = ((rowBase < HALF_B) == (colBase < HALF_B)) ? 1.f : -1.f;
    float fac  = (ti == tj) ? 1.f : 2.f;   // symmetric: count (i,j) and (j,i)
    ws->partial[pidx] = (double)(red[0] * wsym * fac);
  }
}

// ---------------- Kernel 5: deterministic final reduction ---------------------
__global__ __launch_bounds__(256) void k_final(const WS* __restrict__ ws,
                                               float* __restrict__ out) {
  __shared__ double red[256];
  int t = threadIdx.x;
  double s = 0.0;
  for (int i = t; i < NT * NT; i += 256) s += ws->partial[i];
  red[t] = s;
  __syncthreads();
  for (int k = 128; k >= 1; k >>= 1) {
    if (t < k) red[t] += red[t + k];
    __syncthreads();
  }
  if (t == 0) out[0] = (float)(red[0] / ((double)HALF_B * (double)HALF_B));
}

extern "C" void kernel_launch(void* const* d_in, const int* in_sizes, int n_in,
                              void* d_out, int out_size, void* d_ws, size_t ws_size,
                              hipStream_t stream) {
  (void)in_sizes; (void)n_in; (void)out_size; (void)ws_size;
  const float* s0 = (const float*)d_in[0];
  const float* s1 = (const float*)d_in[1];
  const float* t0 = (const float*)d_in[2];
  const float* t1 = (const float*)d_in[3];
  WS* ws = (WS*)d_ws;
  float* out = (float*)d_out;

  k_rownorm<<<NROW / 8, 256, 0, stream>>>(s0, s1, t0, t1, ws);
  k_colsum <<<32, 256, 0, stream>>>(s0, s1, t0, t1, ws);
  k_gamma  <<<1, 256, 0, stream>>>(ws);
  dim3 grid(NT, NT);
  k_mmd    <<<grid, 256, 0, stream>>>(s0, s1, t0, t1, ws);
  k_final  <<<1, 256, 0, stream>>>(ws, out);
}
